// AttnBlock_33071248179420
// MI455X (gfx1250) — matrix operand order repaired and checked by reading, not compiled, not run on hardware
//
#include <hip/hip_runtime.h>
#include <math.h>

// ---------------------------------------------------------------------------
// AttnBlock for MI455X (gfx1250):
//   - all GEMMs via v_wmma_f32_16x16x32_f16 (f16 in, f32 accumulate)
//   - tile staging via Tensor Data Mover (tensor_load_to_lds) with HW LDS
//     padding, double-buffered, tracked on TENSORcnt
// ---------------------------------------------------------------------------

typedef __attribute__((ext_vector_type(16))) _Float16 v16h;
typedef __attribute__((ext_vector_type(8)))  _Float16 v8h;
typedef __attribute__((ext_vector_type(8)))  float    v8f;
typedef __attribute__((ext_vector_type(4)))  unsigned int u32x4;
typedef __attribute__((ext_vector_type(8)))  int          i32x8;
typedef __attribute__((ext_vector_type(4)))  int          i32x4;

#define TPB 256
#define BM  128
#define BN  64
#define BK  32
#define LDPAD 40   // padded LDS row stride in halves (32 data + 8 pad = 80 B)

#define ASZ (BM * LDPAD)     // 5120 halves
#define BSZ (BN * LDPAD)     // 2560 halves
#define BUFSZ (ASZ + BSZ)    // one double-buffer slot (halves)

// Problem constants
#define CC    512
#define NPIX  4096   // 64*64
#define BATCH 4

// ---------------------------------------------------------------------------
// TDM: issue a 2D tile load (tile_rows x 32 f16, row stride = ld elems) into
// LDS at byte offset lds_addr, with HW padding of 8 halves after each 32-half
// row (pad_interval=3 -> every 16 DWORDs, pad_amount=3 -> 4 DWORDs).
// D# layout per CDNA5 ISA 8.3/8.4. 2D tensor: remaining groups zero (NULL).
// ---------------------------------------------------------------------------
__device__ __forceinline__ void tdm_load_tile(unsigned lds_addr,
                                              const _Float16* gptr,
                                              int tile_rows, long long ld) {
  unsigned long long ga = (unsigned long long)(uintptr_t)gptr;
  u32x4 g0;
  g0[0] = 1u;                                      // count=1 (valid), user mode
  g0[1] = lds_addr;                                // LDS byte address
  g0[2] = (unsigned)(ga & 0xFFFFFFFFull);          // global_addr[31:0]
  g0[3] = (unsigned)((ga >> 32) & 0x01FFFFFFull)   // global_addr[56:32]
          | (2u << 30);                            // type=2 (image)

  const unsigned td0 = 1u << 30;                   // tensor_dim0 (OOB bound)
  const unsigned td1 = 1u << 30;                   // tensor_dim1 (OOB bound)
  const unsigned long long st0 = (unsigned long long)ld;  // row stride (elems)

  i32x8 g1;
  // w0: wg_mask=0 | data_size=1(2B)<<16 | pad_enable<<20 | pad_interval=3<<22
  //     | pad_amount=3<<25
  g1[0] = (int)((1u << 16) | (1u << 20) | (3u << 22) | (3u << 25));
  g1[1] = (int)((td0 & 0xFFFFu) << 16);                       // tensor_dim0 lo
  g1[2] = (int)((td0 >> 16) | ((td1 & 0xFFFFu) << 16));       // dim0 hi|dim1 lo
  g1[3] = (int)((td1 >> 16) | (32u << 16));                   // dim1 hi|tile_dim0=32
  g1[4] = (int)(unsigned)tile_rows;                           // tile_dim1 (tile_dim2=0)
  g1[5] = (int)(unsigned)(st0 & 0xFFFFFFFFull);               // dim0_stride lo
  g1[6] = (int)(unsigned)((st0 >> 32) & 0xFFFFull);           // dim0_stride hi
  g1[7] = 0;

  i32x4 z4 = {0, 0, 0, 0};
  i32x8 z8 = {0, 0, 0, 0, 0, 0, 0, 0};
  __builtin_amdgcn_tensor_load_to_lds(g0, g1, z4, z4, z8, 0);
}

// ---------------------------------------------------------------------------
// Fragment loads from LDS following CDNA5 wave32 WMMA VGPR layouts
// A (16x32, MxK): lane<16 -> m=lane, K chunks [h*8..h*8+7] and [16+h*8..+7]
// B (32x16, KxN): lane&15 = n, K chunks [h*8..h*8+7] and [16+h*8..+7]
// C/D (16x16 f32): lane&15 = n, vgpr r -> m = r + 8*(lane>>4)
// ---------------------------------------------------------------------------
__device__ __forceinline__ v16h load_a_frag(const _Float16* sA, int m, int half) {
  v8h lo = *(const v8h*)(sA + m * LDPAD + half * 8);
  v8h hi = *(const v8h*)(sA + m * LDPAD + 16 + half * 8);
  v16h a;
#pragma unroll
  for (int t = 0; t < 8; ++t) { a[t] = lo[t]; a[t + 8] = hi[t]; }
  return a;
}

__device__ __forceinline__ v16h load_b_frag(const _Float16* sB, int n, int half) {
  v8h lo = *(const v8h*)(sB + n * LDPAD + half * 8);
  v8h hi = *(const v8h*)(sB + n * LDPAD + 16 + half * 8);
  v16h b;
#pragma unroll
  for (int t = 0; t < 8; ++t) { b[t] = lo[t]; b[t + 8] = hi[t]; }
  return b;
}

// ---------------------------------------------------------------------------
// Generic GEMM: D[M][N] = A[M][K] * B[N][K]^T  (both operands k-contiguous)
// Epilogue: val = acc*scale + bias[n] (+ resid) ; store normal or transposed.
// Grid: (N/BN, M/BM, batch).  Block: 256 threads = 8 waves (4x2 wave tile).
// ---------------------------------------------------------------------------
template <typename OutT, bool TRANS, bool RESID>
__global__ __launch_bounds__(TPB)
void gemm_wmma(const _Float16* __restrict__ Ab, const _Float16* __restrict__ Bb,
               const float* __restrict__ bias, const float* __restrict__ residb,
               OutT* __restrict__ outb,
               int M, int N, int K, float scale,
               long long strideA, long long strideB, long long strideO,
               long long strideR) {
  __shared__ _Float16 smem[2 * BUFSZ];   // 30720 B, double buffered

  const int z = blockIdx.z;
  const _Float16* A = Ab + (long long)z * strideA;
  const _Float16* B = Bb + (long long)z * strideB;
  OutT* out = outb + (long long)z * strideO;
  const float* resid = RESID ? (residb + (long long)z * strideR) : nullptr;

  const int tid  = threadIdx.x;
  const int lane = tid & 31;
  const int wave = tid >> 5;        // 0..7
  const int half = lane >> 4;       // 0/1
  const int l16  = lane & 15;

  const int mBase = blockIdx.y * BM;
  const int nBase = blockIdx.x * BN;
  const int wm = (wave >> 1) * 32;  // wave row offset in tile
  const int wn = (wave & 1) * 32;   // wave col offset in tile

  const unsigned ldsBase = (unsigned)(unsigned long long)(uintptr_t)(void*)smem;

  v8f acc[2][2];
#pragma unroll
  for (int i = 0; i < 2; ++i)
#pragma unroll
    for (int j = 0; j < 2; ++j)
#pragma unroll
      for (int r = 0; r < 8; ++r) acc[i][j][r] = 0.0f;

  const int T = K / BK;

  // Prologue: wave 0 prefetches tile 0 into buffer 0 via TDM.
  if (wave == 0) {
    tdm_load_tile(ldsBase, A + (long long)mBase * K, BM, K);
    tdm_load_tile(ldsBase + ASZ * 2u, B + (long long)nBase * K, BN, K);
  }

  for (int t = 0; t < T; ++t) {
    const int cur = t & 1;
    const _Float16* sA = smem + cur * BUFSZ;
    const _Float16* sB = sA + ASZ;

    if (wave == 0) {
      if (t + 1 < T) {
        const int nxt = (t + 1) & 1;
        const unsigned nb = ldsBase + (unsigned)(nxt * BUFSZ) * 2u;
        tdm_load_tile(nb, A + (long long)mBase * K + (t + 1) * BK, BM, K);
        tdm_load_tile(nb + ASZ * 2u, B + (long long)nBase * K + (t + 1) * BK, BN, K);
        __builtin_amdgcn_s_wait_tensorcnt(2);  // tile t landed; t+1 in flight
      } else {
        __builtin_amdgcn_s_wait_tensorcnt(0);  // last tile landed
      }
    }
    __syncthreads();   // tile t visible to all waves

    v16h af[2], bf[2];
#pragma unroll
    for (int fm = 0; fm < 2; ++fm) af[fm] = load_a_frag(sA, wm + fm * 16 + l16, half);
#pragma unroll
    for (int fn = 0; fn < 2; ++fn) bf[fn] = load_b_frag(sB, wn + fn * 16 + l16, half);

#pragma unroll
    for (int fm = 0; fm < 2; ++fm)
#pragma unroll
      for (int fn = 0; fn < 2; ++fn)
        acc[fm][fn] = __builtin_amdgcn_wmma_f32_16x16x32_f16(
            false, af[fm], false, bf[fn], (short)0, acc[fm][fn], false, false);

    __syncthreads();   // buffer reuse fence (overwritten at step t+2 issue)
  }

  // Epilogue
#pragma unroll
  for (int fm = 0; fm < 2; ++fm) {
#pragma unroll
    for (int fn = 0; fn < 2; ++fn) {
      const int ng = nBase + wn + fn * 16 + l16;
      const float bv = bias ? bias[ng] : 0.0f;
#pragma unroll
      for (int r = 0; r < 8; ++r) {
        const int mg = mBase + wm + fm * 16 + r + 8 * half;
        float v = acc[fm][fn][r] * scale + bv;
        long long idx = TRANS ? ((long long)ng * M + mg) : ((long long)mg * N + ng);
        if (RESID) v += resid[idx];
        out[idx] = (OutT)v;
      }
    }
  }
}

// ---------------------------------------------------------------------------
// Weight f32 -> f16 conversion
// ---------------------------------------------------------------------------
__global__ __launch_bounds__(TPB)
void cvt_f16(const float* __restrict__ w, _Float16* __restrict__ o, int n) {
  int i = blockIdx.x * TPB + threadIdx.x;
  if (i < n) o[i] = (_Float16)w[i];
}

// ---------------------------------------------------------------------------
// GroupNorm stats: one block per (batch, group); group data is contiguous
// (32 channels * 4096 pixels = 131072 floats)
// ---------------------------------------------------------------------------
__global__ __launch_bounds__(TPB)
void gn_stats(const float* __restrict__ x, float* __restrict__ stats) {
  const int bg = blockIdx.x;  // 0..63
  const float* p = x + (long long)bg * 131072;
  float s = 0.0f, sq = 0.0f;
  for (int i = threadIdx.x; i < 131072; i += TPB) {
    float v = p[i];
    s += v;
    sq += v * v;
  }
  __shared__ float rs[TPB], rq[TPB];
  rs[threadIdx.x] = s;
  rq[threadIdx.x] = sq;
  __syncthreads();
  for (int off = TPB / 2; off > 0; off >>= 1) {
    if (threadIdx.x < off) {
      rs[threadIdx.x] += rs[threadIdx.x + off];
      rq[threadIdx.x] += rq[threadIdx.x + off];
    }
    __syncthreads();
  }
  if (threadIdx.x == 0) {
    float mean = rs[0] * (1.0f / 131072.0f);
    float var  = rq[0] * (1.0f / 131072.0f) - mean * mean;
    stats[2 * bg]     = mean;
    stats[2 * bg + 1] = rsqrtf(var + 1e-6f);
  }
}

// ---------------------------------------------------------------------------
// GroupNorm apply + transpose to hn[b][i][c] (f16), GEMM-friendly layout
// ---------------------------------------------------------------------------
__global__ __launch_bounds__(TPB)
void gn_apply(const float* __restrict__ x, const float* __restrict__ gamma,
              const float* __restrict__ beta, const float* __restrict__ stats,
              _Float16* __restrict__ hn) {
  long long idx = (long long)blockIdx.x * TPB + threadIdx.x;  // < 4*512*4096
  int i = (int)(idx & 4095);
  int c = (int)((idx >> 12) & 511);
  int b = (int)(idx >> 21);
  int bg = b * 16 + (c >> 5);
  float v = (x[idx] - stats[2 * bg]) * stats[2 * bg + 1] * gamma[c] + beta[c];
  hn[(((long long)b * NPIX) + i) * CC + c] = (_Float16)v;
}

// ---------------------------------------------------------------------------
// Row softmax over 4096 f16 scores, in place; f32 math.
// ---------------------------------------------------------------------------
__global__ __launch_bounds__(TPB)
void softmax_rows(_Float16* __restrict__ attn) {
  const long long base = (long long)blockIdx.x * NPIX;
  const int tid = threadIdx.x;
  float vals[16];
  float m = -3.0e38f;
#pragma unroll
  for (int j = 0; j < 16; ++j) {
    float v = (float)attn[base + tid + j * TPB];
    vals[j] = v;
    m = fmaxf(m, v);
  }
  __shared__ float red[TPB];
  red[tid] = m;
  __syncthreads();
  for (int off = TPB / 2; off > 0; off >>= 1) {
    if (tid < off) red[tid] = fmaxf(red[tid], red[tid + off]);
    __syncthreads();
  }
  const float rmax = red[0];
  __syncthreads();
  float s = 0.0f;
#pragma unroll
  for (int j = 0; j < 16; ++j) {
    float e = __expf(vals[j] - rmax);
    vals[j] = e;
    s += e;
  }
  red[tid] = s;
  __syncthreads();
  for (int off = TPB / 2; off > 0; off >>= 1) {
    if (tid < off) red[tid] += red[tid + off];
    __syncthreads();
  }
  const float inv = 1.0f / red[0];
#pragma unroll
  for (int j = 0; j < 16; ++j) attn[base + tid + j * TPB] = (_Float16)(vals[j] * inv);
}

// ---------------------------------------------------------------------------
// Host-side orchestration
// ---------------------------------------------------------------------------
extern "C" void kernel_launch(void* const* d_in, const int* in_sizes, int n_in,
                              void* d_out, int out_size, void* d_ws, size_t ws_size,
                              hipStream_t stream) {
  const float* x     = (const float*)d_in[0];
  const float* gamma = (const float*)d_in[1];
  const float* beta  = (const float*)d_in[2];
  const float* wq    = (const float*)d_in[3];
  const float* bq    = (const float*)d_in[4];
  const float* wk    = (const float*)d_in[5];
  const float* bk    = (const float*)d_in[6];
  const float* wv    = (const float*)d_in[7];
  const float* bv    = (const float*)d_in[8];
  const float* wp    = (const float*)d_in[9];
  const float* bp    = (const float*)d_in[10];
  float* out = (float*)d_out;

  char* ws = (char*)d_ws;
  const long long WSZ  = (long long)CC * CC;             // elems per weight
  const long long HNSZ = (long long)BATCH * NPIX * CC;   // f16 elems
  const long long ATSZ = (long long)BATCH * NPIX * NPIX; // f16 elems
  (void)ATSZ;

  _Float16* wq16  = (_Float16*)(ws);
  _Float16* wk16  = wq16 + WSZ;
  _Float16* wv16  = wk16 + WSZ;
  _Float16* wp16  = wv16 + WSZ;
  float*    stats = (float*)(ws + (4 * WSZ) * sizeof(_Float16));   // 128 floats
  _Float16* hn    = (_Float16*)(ws + 4 * 1024 * 1024);             // 16 MB
  _Float16* qbuf  = hn + HNSZ;                                     // [b][i][c]
  _Float16* kbuf  = qbuf + HNSZ;                                   // [b][j][c]
  _Float16* vbuf  = kbuf + HNSZ;                                   // [b][c][j]
  _Float16* aobuf = vbuf + HNSZ;                                   // [b][i][c]
  _Float16* attn  = aobuf + HNSZ;                                  // [b][i][j]

  // 1) weights -> f16
  cvt_f16<<<(int)((WSZ + TPB - 1) / TPB), TPB, 0, stream>>>(wq, wq16, (int)WSZ);
  cvt_f16<<<(int)((WSZ + TPB - 1) / TPB), TPB, 0, stream>>>(wk, wk16, (int)WSZ);
  cvt_f16<<<(int)((WSZ + TPB - 1) / TPB), TPB, 0, stream>>>(wv, wv16, (int)WSZ);
  cvt_f16<<<(int)((WSZ + TPB - 1) / TPB), TPB, 0, stream>>>(wp, wp16, (int)WSZ);

  // 2) GroupNorm
  gn_stats<<<BATCH * 16, TPB, 0, stream>>>(x, stats);
  gn_apply<<<(int)(((long long)BATCH * CC * NPIX) / TPB), TPB, 0, stream>>>(
      x, gamma, beta, stats, hn);

  const long long sHN = (long long)NPIX * CC;
  const long long sAT = (long long)NPIX * NPIX;

  // 3) Q/K/V projections
  {
    dim3 g(CC / BN, NPIX / BM, BATCH);
    gemm_wmma<_Float16, false, false><<<g, TPB, 0, stream>>>(
        hn, wq16, bq, nullptr, qbuf, NPIX, CC, CC, 1.0f, sHN, 0, sHN, 0);
    gemm_wmma<_Float16, false, false><<<g, TPB, 0, stream>>>(
        hn, wk16, bk, nullptr, kbuf, NPIX, CC, CC, 1.0f, sHN, 0, sHN, 0);
    // V stored transposed: v[b][c][j]
    gemm_wmma<_Float16, true, false><<<g, TPB, 0, stream>>>(
        hn, wv16, bv, nullptr, vbuf, NPIX, CC, CC, 1.0f, sHN, 0, sHN, 0);
  }

  // 4) scores = (Q K^T) / sqrt(c)  -> attn[b][i][j] f16
  {
    dim3 g(NPIX / BN, NPIX / BM, BATCH);
    const float scale = 1.0f / sqrtf((float)CC);
    gemm_wmma<_Float16, false, false><<<g, TPB, 0, stream>>>(
        qbuf, kbuf, nullptr, nullptr, attn, NPIX, NPIX, CC, scale, sHN, sHN, sAT, 0);
  }

  // 5) softmax over keys j, in place
  softmax_rows<<<BATCH * NPIX, TPB, 0, stream>>>(attn);

  // 6) attnout[b][i][c] = attn[b][i][:] . v[b][c][:]
  {
    dim3 g(CC / BN, NPIX / BM, BATCH);
    gemm_wmma<_Float16, false, false><<<g, TPB, 0, stream>>>(
        attn, vbuf, nullptr, nullptr, aobuf, NPIX, CC, NPIX, 1.0f, sAT, sHN, sHN, 0);
  }

  // 7) out[b][c][i] = x + attnout * Wp^T + bp   (transposed store + residual)
  {
    dim3 g(CC / BN, NPIX / BM, BATCH);
    gemm_wmma<float, true, true><<<g, TPB, 0, stream>>>(
        aobuf, wp16, bp, x, out, NPIX, CC, CC, 1.0f, sHN, 0, sHN, sHN);
  }
}
